// LSTMEDModule_19464791785504
// MI455X (gfx1250) — matrix operand order repaired and checked by reading, not compiled, not run on hardware
//
#include <hip/hip_runtime.h>
#include <math.h>

typedef _Float16 f16;
typedef __attribute__((ext_vector_type(16))) _Float16 v16h;
typedef __attribute__((ext_vector_type(8)))  float    v8f;

#define BB 512
#define SS 512
#define FF 32
#define HH 256
#define GG 1024   // 4*H

// ---------------- fragment index helpers ----------------
// A-matrix (16-bit, 16x32): lanes 0-15 hold K {0..7,16..23}, lanes 16-31 hold K {8..15,24..31}
__device__ __host__ inline int a_k_from(int laneHalf, int e) {
    return ((e & 8) ? 16 : 0) + laneHalf * 8 + (e & 7);
}
// inverse: K -> (laneHalf, e)
__device__ inline void a_inv(int K, int& laneHalf, int& e) {
    laneHalf = ((K & 15) >= 8) ? 1 : 0;
    e = ((K >= 16) ? 8 : 0) + (K & 7);
}
// B-matrix (16-bit, 32x16): lane holds column n = lane%16; K = ((e&8)?16:0) + laneHalf*8 + (e&7)
__device__ inline int b_k_from(int laneHalf, int e) {
    return ((e & 8) ? 16 : 0) + laneHalf * 8 + (e & 7);
}

// ---------------- prep: pack x (all timesteps) into A-fragment layout ----------------
// dst layout: [s][rt(32)][lane(32)][e(16)]  halves
__global__ void pack_x_kernel(const float* __restrict__ ts, f16* __restrict__ dst, int total) {
    int idx = blockIdx.x * blockDim.x + threadIdx.x;
    if (idx >= total) return;
    int e    = idx & 15;
    int lane = (idx >> 4) & 31;
    int rest = idx >> 9;
    int rt   = rest & 31;
    int s    = rest >> 5;
    int m = lane & 15, laneHalf = lane >> 4;
    int K = a_k_from(laneHalf, e);          // K == feature index (F=32)
    int b = rt * 16 + m;
    dst[idx] = (f16)ts[((size_t)b * SS + s) * FF + K];
}

// ---------------- prep: pack weight W[N,K] (row-major f32) into B-fragment layout ------
// dst layout: [nt(N/16)][kt(K/32)][lane(32)][e(16)] halves
__global__ void pack_w_kernel(const float* __restrict__ W, f16* __restrict__ dst,
                              int N, int K, int total) {
    int idx = blockIdx.x * blockDim.x + threadIdx.x;
    if (idx >= total) return;
    int e    = idx & 15;
    int lane = (idx >> 4) & 31;
    int rest = idx >> 9;
    int KT   = K >> 5;
    int kt   = rest % KT;
    int nt   = rest / KT;
    int n = lane & 15, laneHalf = lane >> 4;
    int col = kt * 32 + b_k_from(laneHalf, e);
    int row = nt * 16 + n;
    dst[idx] = (f16)W[(size_t)row * K + col];
}

// ---------------- prep: zero h (packed) and c ----------------
__global__ void init_state_kernel(f16* __restrict__ h0, float* __restrict__ c, int total) {
    int idx = blockIdx.x * blockDim.x + threadIdx.x;
    if (idx >= total) return;
    h0[idx] = (f16)0.f;
    c[idx] = 0.f;
}

// ---------------- fused LSTM step (+ optional fused output projection) ----------------
// grid (32 row-tiles, 16 col-tiles of H), block = 128 (4 waves; wave w = gate w)
// If yOut != nullptr, blocks with ct < 2 also compute y = h_old @ W_out.T + b_out
// for output time tOut (proj tile nt == ct), fused into the same barrier phase.
__global__ __launch_bounds__(128) void lstm_step_kernel(
    const f16* __restrict__ xP,     // [32][32][16]  A-frag of x_t
    const f16* __restrict__ hR,     // [32][8][32][16] A-frag of h (read)
    f16* __restrict__ hW,           // same layout (write)
    float* __restrict__ cBuf,       // [512][256] f32
    const f16* __restrict__ WihP,   // [64][1][32][16] B-frag
    const f16* __restrict__ WhhP,   // [64][8][32][16] B-frag
    const float* __restrict__ bih, const float* __restrict__ bhh,
    const f16* __restrict__ WoutP,  // [2][8][32][16] B-frag (may be unused)
    const float* __restrict__ b_out,
    float* __restrict__ yOut, int tOut)
{
    __shared__ float zTile[4][256];
    __shared__ float pTile[4][256];
    int rt = blockIdx.x;            // row tile 0..31
    int ct = blockIdx.y;            // H column tile 0..15
    int w    = threadIdx.x >> 5;    // gate: 0=i 1=f 2=g 3=o
    int lane = threadIdx.x & 31;
    int nt = w * 16 + ct;           // column tile in 4H space
    bool doProj = (yOut != nullptr) && (ct < 2);   // block-uniform

    v8f acc = {};
    // x @ Wih.T  (single K-tile, K = F = 32)
    {
        v16h a = *(const v16h*)(xP + (((size_t)rt * 32 + lane) << 4));
        v16h b = *(const v16h*)(WihP + (((size_t)nt * 32 + lane) << 4));
        acc = __builtin_amdgcn_wmma_f32_16x16x32_f16(false, a, false, b,
                                                     (short)0, acc, false, false);
    }
    // h @ Whh.T  (8 K-tiles over H=256)
#pragma unroll
    for (int kt = 0; kt < 8; ++kt) {
        v16h a = *(const v16h*)(hR + ((((size_t)rt * 8 + kt) * 32 + lane) << 4));
        v16h b = *(const v16h*)(WhhP + ((((size_t)nt * 8 + kt) * 32 + lane) << 4));
        acc = __builtin_amdgcn_wmma_f32_16x16x32_f16(false, a, false, b,
                                                     (short)0, acc, false, false);
    }

    int n = lane & 15;
    int mbase = (lane < 16) ? 0 : 8;          // C/D layout: VGPR r -> row mbase+r, col n

    // fused projection partials: wave w does k-tiles {2w, 2w+1} of h_old @ W_out.T
    if (doProj) {
        v8f pacc = {};
#pragma unroll
        for (int kk = 0; kk < 2; ++kk) {
            int kt = w * 2 + kk;
            v16h a = *(const v16h*)(hR + ((((size_t)rt * 8 + kt) * 32 + lane) << 4));
            v16h b = *(const v16h*)(WoutP + ((((size_t)ct * 8 + kt) * 32 + lane) << 4));
            pacc = __builtin_amdgcn_wmma_f32_16x16x32_f16(false, a, false, b,
                                                          (short)0, pacc, false, false);
        }
#pragma unroll
        for (int r = 0; r < 8; ++r)
            pTile[w][(mbase + r) * 16 + n] = pacc[r];
    }

    int j = nt * 16 + n;                      // column in 4H
    float bval = bih[j] + bhh[j];
#pragma unroll
    for (int r = 0; r < 8; ++r)
        zTile[w][(mbase + r) * 16 + n] = acc[r] + bval;
    __syncthreads();

    // fuse gates (and projection reduction): 256 elements over 128 threads
    for (int idx = threadIdx.x; idx < 256; idx += 128) {
        int m = idx >> 4, nn = idx & 15;
        float iv = zTile[0][idx], fv = zTile[1][idx];
        float gv = zTile[2][idx], ov = zTile[3][idx];
        float si = 1.f / (1.f + __expf(-iv));
        float sf = 1.f / (1.f + __expf(-fv));
        float so = 1.f / (1.f + __expf(-ov));
        float tg = tanhf(gv);
        int brow = rt * 16 + m;
        int jh   = ct * 16 + nn;
        float cold = cBuf[(size_t)brow * HH + jh];
        float cnew = sf * cold + si * tg;
        cBuf[(size_t)brow * HH + jh] = cnew;
        float hnew = so * tanhf(cnew);
        // scatter h_new into A-fragment layout for next step
        int kt = jh >> 5, K = jh & 31;
        int laneHalf, e;
        a_inv(K, laneHalf, e);
        int plane = m + laneHalf * 16;
        hW[((((size_t)rt * 8 + kt) * 32 + plane) << 4) + e] = (f16)hnew;

        if (doProj) {
            int fcol = ct * 16 + nn;          // F column (nt == ct, F=32)
            float y = pTile[0][idx] + pTile[1][idx] + pTile[2][idx] + pTile[3][idx]
                    + b_out[fcol];
            yOut[((size_t)brow * SS + tOut) * FF + fcol] = y;
        }
    }
}

extern "C" void kernel_launch(void* const* d_in, const int* in_sizes, int n_in,
                              void* d_out, int out_size, void* d_ws, size_t ws_size,
                              hipStream_t stream) {
    (void)in_sizes; (void)n_in; (void)out_size; (void)ws_size;
    const float* ts      = (const float*)d_in[0];
    const float* enc_Wih = (const float*)d_in[1];
    const float* enc_Whh = (const float*)d_in[2];
    const float* enc_bih = (const float*)d_in[3];
    const float* enc_bhh = (const float*)d_in[4];
    const float* dec_Wih = (const float*)d_in[5];
    const float* dec_Whh = (const float*)d_in[6];
    const float* dec_bih = (const float*)d_in[7];
    const float* dec_bhh = (const float*)d_in[8];
    const float* W_out   = (const float*)d_in[9];
    const float* b_out   = (const float*)d_in[10];
    float* out = (float*)d_out;

    // workspace carve-up (all offsets 32B-aligned)
    f16* XP    = (f16*)d_ws;                  // 512 * 16384 halves
    f16* H0    = XP + (size_t)SS * 16384;     // 131072 halves
    f16* H1    = H0 + 131072;                 // 131072
    f16* WhhE  = H1 + 131072;                 // 262144
    f16* WhhD  = WhhE + 262144;               // 262144
    f16* WihE  = WhhD + 262144;               // 32768
    f16* WihD  = WihE + 32768;                // 32768
    f16* WoutP = WihD + 32768;                // 8192
    float* C   = (float*)(WoutP + 8192);      // 131072 floats
    f16* hbuf[2] = {H0, H1};

    // ---- prep ----
    {
        int total = SS * 32 * 32 * 16; // 8,388,608
        pack_x_kernel<<<(total + 255) / 256, 256, 0, stream>>>(ts, XP, total);
    }
    pack_w_kernel<<<(GG * HH + 255) / 256, 256, 0, stream>>>(enc_Whh, WhhE, GG, HH, GG * HH);
    pack_w_kernel<<<(GG * HH + 255) / 256, 256, 0, stream>>>(dec_Whh, WhhD, GG, HH, GG * HH);
    pack_w_kernel<<<(GG * FF + 255) / 256, 256, 0, stream>>>(enc_Wih, WihE, GG, FF, GG * FF);
    pack_w_kernel<<<(GG * FF + 255) / 256, 256, 0, stream>>>(dec_Wih, WihD, GG, FF, GG * FF);
    pack_w_kernel<<<(FF * HH + 255) / 256, 256, 0, stream>>>(W_out, WoutP, FF, HH, FF * HH);
    init_state_kernel<<<(131072 + 255) / 256, 256, 0, stream>>>(H0, C, 131072);

    dim3 stepGrid(32, 16), stepBlock(128);

    // ---- encoder: t = 0..S-1 (no projection) ----
    for (int t = 0; t < SS; ++t) {
        lstm_step_kernel<<<stepGrid, stepBlock, 0, stream>>>(
            XP + (size_t)t * 16384, hbuf[t & 1], hbuf[(t + 1) & 1], C,
            WihE, WhhE, enc_bih, enc_bhh,
            WoutP, b_out, nullptr, 0);
    }
    // latent = c_enc : contiguous [B,H] f32
    hipMemcpyAsync(out + (size_t)BB * SS * FF, C, (size_t)BB * HH * sizeof(float),
                   hipMemcpyDeviceToDevice, stream);

    // ---- decoder: step t fuses y-projection (from h BEFORE the step) at time S-1-t ----
    for (int t = 0; t < SS; ++t) {
        int u = SS + t;                 // global parity index (h_enc lives in hbuf[0])
        lstm_step_kernel<<<stepGrid, stepBlock, 0, stream>>>(
            XP + (size_t)(SS - 1 - t) * 16384, hbuf[u & 1], hbuf[(u + 1) & 1], C,
            WihD, WhhD, dec_bih, dec_bhh,
            WoutP, b_out, out, SS - 1 - t);
    }
}
